// ControlCPC_76940044140864
// MI455X (gfx1250) — matrix operand order repaired and checked by reading, not compiled, not run on hardware
//
#include <hip/hip_runtime.h>
#include <math.h>

typedef __attribute__((ext_vector_type(16))) __bf16 v16bf;
typedef __attribute__((ext_vector_type(8)))  float  v8f;

#define B_SZ   1024
#define Z_DIM  512
#define A_DIM  8
#define AH_DIM 64
#define NSHIFT 99      /* n-1 */
#define NCOL   1123    /* B + n - 1 */
#define INV_T  10.0f   /* 1 / temperature */

static __device__ __forceinline__ void bf16_split(float v, __bf16& hi, __bf16& lo) {
  hi = (__bf16)v;
  lo = (__bf16)(v - (float)hi);
}

// ---------------------------------------------------------------------------
// h_a = relu(actions @ Wa + ba)   [1024, 64] fp32 (A_DIM=8, trivial)
// ---------------------------------------------------------------------------
__global__ void ha_kernel(const float* __restrict__ actions, const float* __restrict__ Wa,
                          const float* __restrict__ ba, float* __restrict__ ha) {
  int t = blockIdx.x * blockDim.x + threadIdx.x;
  if (t >= B_SZ * AH_DIM) return;
  int b = t >> 6, j = t & 63;
  float s = ba[j];
  #pragma unroll
  for (int k = 0; k < A_DIM; ++k)
    s = fmaf(actions[b * A_DIM + k], Wa[k * AH_DIM + j], s);
  ha[t] = fmaxf(s, 0.0f);
}

// ---------------------------------------------------------------------------
// Pack row-major fp32 A [Mt*16, K] into WMMA 16x32 bf16 A-fragments (hi + lo).
// Fragment layout (this code's contiguous-k order, wave32): lane l holds row M=l&15;
// dword j holds K = 16*(l>>4) + 2j and K+1.
// ---------------------------------------------------------------------------
__global__ void pack_a_kernel(const float* __restrict__ src, __bf16* __restrict__ dh,
                              __bf16* __restrict__ dl, int Mt, int KT, int K) {
  int t = blockIdx.x * blockDim.x + threadIdx.x;
  if (t >= Mt * KT * 32) return;
  int lane = t & 31, f = t >> 5;
  int kt = f % KT, mt = f / KT;
  int m = mt * 16 + (lane & 15);
  size_t base = (size_t)f * 512 + (size_t)lane * 16;
  #pragma unroll
  for (int j = 0; j < 8; ++j) {
    int k = kt * 32 + 16 * (lane >> 4) + 2 * j;
    __bf16 h0, l0, h1, l1;
    bf16_split(src[(size_t)m * K + k],     h0, l0);
    bf16_split(src[(size_t)m * K + k + 1], h1, l1);
    dh[base + 2 * j] = h0; dh[base + 2 * j + 1] = h1;
    dl[base + 2 * j] = l0; dl[base + 2 * j + 1] = l1;
  }
}

// ---------------------------------------------------------------------------
// Pack fp32 B (row-major [K,N], or transposed source [N,K] when transB=1)
// into WMMA 32x16 bf16 B-fragments (hi + lo).
// Layout: lane l holds column N=l&15; dword j holds K = 16*(l>>4) + 2j, +1.
// ---------------------------------------------------------------------------
__global__ void pack_b_kernel(const float* __restrict__ src, __bf16* __restrict__ dh,
                              __bf16* __restrict__ dl, int KT, int Nt, int ldb, int transB) {
  int t = blockIdx.x * blockDim.x + threadIdx.x;
  if (t >= KT * Nt * 32) return;
  int lane = t & 31, f = t >> 5;
  int nt = f % Nt, kt = f / Nt;
  int n = nt * 16 + (lane & 15);
  size_t base = (size_t)f * 512 + (size_t)lane * 16;
  #pragma unroll
  for (int j = 0; j < 8; ++j) {
    int k = kt * 32 + 16 * (lane >> 4) + 2 * j;
    size_t i0 = transB ? ((size_t)n * ldb + k)     : ((size_t)k * ldb + n);
    size_t i1 = transB ? ((size_t)n * ldb + k + 1) : ((size_t)(k + 1) * ldb + n);
    __bf16 h0, l0, h1, l1;
    bf16_split(src[i0], h0, l0);
    bf16_split(src[i1], h1, l1);
    dh[base + 2 * j] = h0; dh[base + 2 * j + 1] = h1;
    dl[base + 2 * j] = l0; dl[base + 2 * j + 1] = l1;
  }
}

// ---------------------------------------------------------------------------
// GEMM on pre-packed bf16 fragments with hi/lo error compensation:
//   C += Ah*Bh + Ah*Bl + Al*Bh   (fp32 accumulate, ~17-bit effective mantissa)
// One wave computes a 16 x 64 output tile (1 Mtile x 4 Ntiles, A reused).
// ---------------------------------------------------------------------------
__global__ void gemm_wmma_kernel(const __bf16* __restrict__ Ah, const __bf16* __restrict__ Al,
                                 const __bf16* __restrict__ Bh, const __bf16* __restrict__ Bl,
                                 float* __restrict__ out, const float* __restrict__ bias,
                                 int KT, int Nt, int ldOut) {
  int lane = threadIdx.x & 31;
  int task = (blockIdx.x * blockDim.x + threadIdx.x) >> 5;
  int ntGroups = Nt >> 2;
  int mt = task / ntGroups;
  int ntg = task % ntGroups;

  v8f c[4] = {};
  for (int kt = 0; kt < KT; ++kt) {
    size_t afo = ((size_t)(mt * KT + kt) * 32 + lane) * 16;
    v16bf ah = *(const v16bf*)(Ah + afo);
    v16bf al = *(const v16bf*)(Al + afo);
    #pragma unroll
    for (int t = 0; t < 4; ++t) {
      int nt = ntg * 4 + t;
      size_t bfo = ((size_t)(kt * Nt + nt) * 32 + lane) * 16;
      v16bf bh = *(const v16bf*)(Bh + bfo);
      v16bf bl = *(const v16bf*)(Bl + bfo);
      c[t] = __builtin_amdgcn_wmma_f32_16x16x32_bf16(false, ah, false, bh, (short)0, c[t], false, false);
      c[t] = __builtin_amdgcn_wmma_f32_16x16x32_bf16(false, ah, false, bl, (short)0, c[t], false, false);
      c[t] = __builtin_amdgcn_wmma_f32_16x16x32_bf16(false, al, false, bh, (short)0, c[t], false, false);
    }
  }
  // C/D layout: lanes 0-15: VGPR v -> (M=v, N=lane); lanes 16-31: (M=v+8, N=lane-16)
  int mrow = (lane >> 4) * 8;
  int ncol = lane & 15;
  #pragma unroll
  for (int t = 0; t < 4; ++t) {
    int n = (ntg * 4 + t) * 16 + ncol;
    float badd = bias ? bias[n] : 0.0f;
    #pragma unroll
    for (int v = 0; v < 8; ++v) {
      int m = mt * 16 + mrow + v;
      out[(size_t)m * ldOut + n] = c[t][v] + badd;
    }
  }
}

// ---------------------------------------------------------------------------
// Negative similarities. One wave per batch row b:
//   base    = sum_n (z[b,n] + b2[n]) * zn[b,n]
//   sim[b, 1024+i-1] = base + sum_n relu(U[b,n] + G[(b+i)%B, n]) * V[b,n]
// U already contains +b1. U,V kept in registers; G streamed from L2.
// ---------------------------------------------------------------------------
__global__ void neg_kernel(const float* __restrict__ U, const float* __restrict__ G,
                           const float* __restrict__ V, const float* __restrict__ z,
                           const float* __restrict__ zn, const float* __restrict__ b2,
                           float* __restrict__ simmat) {
  int lane = threadIdx.x & 31;
  int b = blockIdx.x * (blockDim.x >> 5) + (threadIdx.x >> 5);
  const float4* Ub = (const float4*)(U + (size_t)b * Z_DIM);
  const float4* Vb = (const float4*)(V + (size_t)b * Z_DIM);
  const float4* Zb = (const float4*)(z + (size_t)b * Z_DIM);
  const float4* Nb = (const float4*)(zn + (size_t)b * Z_DIM);
  const float4* B2 = (const float4*)b2;

  float4 u[4], v[4];
  float base = 0.0f;
  #pragma unroll
  for (int t = 0; t < 4; ++t) {
    int idx = lane * 4 + t;
    u[t] = Ub[idx];
    v[t] = Vb[idx];
    float4 zz = Zb[idx], nn = Nb[idx], bb = B2[idx];
    base += (zz.x + bb.x) * nn.x + (zz.y + bb.y) * nn.y +
            (zz.z + bb.z) * nn.z + (zz.w + bb.w) * nn.w;
  }
  #pragma unroll
  for (int m = 16; m >= 1; m >>= 1) base += __shfl_xor(base, m);

  for (int i = 1; i <= NSHIFT; ++i) {
    const float* Grow = G + (size_t)((b + i) & (B_SZ - 1)) * Z_DIM;
    __builtin_prefetch(G + (size_t)((b + i + 1) & (B_SZ - 1)) * Z_DIM + lane * 16, 0, 0);
    const float4* Gb = (const float4*)Grow;
    float s = 0.0f;
    #pragma unroll
    for (int t = 0; t < 4; ++t) {
      float4 g = Gb[lane * 4 + t];
      s = fmaf(fmaxf(u[t].x + g.x, 0.0f), v[t].x, s);
      s = fmaf(fmaxf(u[t].y + g.y, 0.0f), v[t].y, s);
      s = fmaf(fmaxf(u[t].z + g.z, 0.0f), v[t].z, s);
      s = fmaf(fmaxf(u[t].w + g.w, 0.0f), v[t].w, s);
    }
    #pragma unroll
    for (int m = 16; m >= 1; m >>= 1) s += __shfl_xor(s, m);
    if (lane == 0) simmat[(size_t)b * NCOL + B_SZ + (i - 1)] = s + base;
  }
}

// ---------------------------------------------------------------------------
// Per-row softmax diag log-prob + rank of the diagonal. One wave per row.
// ---------------------------------------------------------------------------
__global__ void rowstat_kernel(const float* __restrict__ simmat,
                               float* __restrict__ row_loss, int* __restrict__ row_rank) {
  int lane = threadIdx.x & 31;
  int b = blockIdx.x * (blockDim.x >> 5) + (threadIdx.x >> 5);
  const float* row = simmat + (size_t)b * NCOL;
  float diag = row[b];

  float mx = -3.0e38f;
  for (int i = lane; i < NCOL; i += 32) mx = fmaxf(mx, row[i]);
  #pragma unroll
  for (int m = 16; m >= 1; m >>= 1) mx = fmaxf(mx, __shfl_xor(mx, m));

  float se = 0.0f;
  int cnt = 0;
  for (int i = lane; i < NCOL; i += 32) {
    float s = row[i];
    se += expf((s - mx) * INV_T);
    cnt += (s > diag) ? 1 : 0;
  }
  #pragma unroll
  for (int m = 16; m >= 1; m >>= 1) { se += __shfl_xor(se, m); cnt += __shfl_xor(cnt, m); }

  if (lane == 0) {
    row_loss[b] = -((diag - mx) * INV_T - logf(se));
    row_rank[b] = cnt;
  }
}

// ---------------------------------------------------------------------------
// Deterministic final reduction: loss mean + top-{1,3,10} accuracies.
// ---------------------------------------------------------------------------
__global__ void reduce_kernel(const float* __restrict__ row_loss, const int* __restrict__ row_rank,
                              float* __restrict__ out) {
  __shared__ float sl[256];
  __shared__ int s1[256], s3[256], s10[256];
  int t = threadIdx.x;
  float l = 0.0f; int c1 = 0, c3 = 0, c10 = 0;
  for (int r = t; r < B_SZ; r += 256) {
    l += row_loss[r];
    int rk = row_rank[r];
    c1 += (rk < 1); c3 += (rk < 3); c10 += (rk < 10);
  }
  sl[t] = l; s1[t] = c1; s3[t] = c3; s10[t] = c10;
  __syncthreads();
  for (int st = 128; st >= 1; st >>= 1) {
    if (t < st) { sl[t] += sl[t + st]; s1[t] += s1[t + st]; s3[t] += s3[t + st]; s10[t] += s10[t + st]; }
    __syncthreads();
  }
  if (t == 0) {
    out[0] = sl[0] / (float)B_SZ;
    out[1] = (float)s1[0] / (float)B_SZ;
    out[2] = (float)s3[0] / (float)B_SZ;
    out[3] = (float)s10[0] / (float)B_SZ;
  }
}

// ---------------------------------------------------------------------------
extern "C" void kernel_launch(void* const* d_in, const int* in_sizes, int n_in,
                              void* d_out, int out_size, void* d_ws, size_t ws_size,
                              hipStream_t stream) {
  const float* z       = (const float*)d_in[0];
  const float* z_next  = (const float*)d_in[1];
  const float* z_hat   = (const float*)d_in[2];
  const float* actions = (const float*)d_in[3];
  const float* Wa      = (const float*)d_in[4];
  const float* ba      = (const float*)d_in[5];
  const float* W1      = (const float*)d_in[6];
  const float* b1      = (const float*)d_in[7];
  const float* W2      = (const float*)d_in[8];
  const float* b2      = (const float*)d_in[9];
  float* out = (float*)d_out;

  char* w = (char*)d_ws;
  size_t off = 0;
  auto nxt = [&](size_t bytes) -> void* {
    off = (off + 255) & ~(size_t)255;
    void* p = w + off;
    off += bytes;
    return p;
  };

  float* ha      = (float*)nxt((size_t)B_SZ * AH_DIM * 4);
  float* U       = (float*)nxt((size_t)B_SZ * Z_DIM * 4);
  float* G       = (float*)nxt((size_t)B_SZ * Z_DIM * 4);
  float* V       = (float*)nxt((size_t)B_SZ * Z_DIM * 4);
  float* simmat  = (float*)nxt((size_t)B_SZ * NCOL * 4);
  __bf16* zPh    = (__bf16*)nxt((size_t)B_SZ * Z_DIM * 2);
  __bf16* zPl    = (__bf16*)nxt((size_t)B_SZ * Z_DIM * 2);
  __bf16* znPh   = (__bf16*)nxt((size_t)B_SZ * Z_DIM * 2);
  __bf16* znPl   = (__bf16*)nxt((size_t)B_SZ * Z_DIM * 2);
  __bf16* haPh   = (__bf16*)nxt((size_t)B_SZ * AH_DIM * 2);
  __bf16* haPl   = (__bf16*)nxt((size_t)B_SZ * AH_DIM * 2);
  __bf16* W1aPh  = (__bf16*)nxt((size_t)Z_DIM * Z_DIM * 2);
  __bf16* W1aPl  = (__bf16*)nxt((size_t)Z_DIM * Z_DIM * 2);
  __bf16* W1bPh  = (__bf16*)nxt((size_t)AH_DIM * Z_DIM * 2);
  __bf16* W1bPl  = (__bf16*)nxt((size_t)AH_DIM * Z_DIM * 2);
  __bf16* W2tPh  = (__bf16*)nxt((size_t)Z_DIM * Z_DIM * 2);
  __bf16* W2tPl  = (__bf16*)nxt((size_t)Z_DIM * Z_DIM * 2);
  __bf16* zhPh   = (__bf16*)nxt((size_t)B_SZ * Z_DIM * 2);
  __bf16* zhPl   = (__bf16*)nxt((size_t)B_SZ * Z_DIM * 2);
  float* row_loss = (float*)nxt((size_t)B_SZ * 4);
  int*   row_rank = (int*)nxt((size_t)B_SZ * 4);

  dim3 blk(256);

  // action embedding (fp32, tiny)
  ha_kernel<<<(B_SZ * AH_DIM + 255) / 256, blk, 0, stream>>>(actions, Wa, ba, ha);

  // pack operands into WMMA fragment layouts (hi/lo bf16 split)
  pack_a_kernel<<<(64 * 16 * 32) / 256, blk, 0, stream>>>(z,      zPh,  zPl,  64, 16, Z_DIM);
  pack_a_kernel<<<(64 * 16 * 32) / 256, blk, 0, stream>>>(z_next, znPh, znPl, 64, 16, Z_DIM);
  pack_a_kernel<<<(64 *  2 * 32) / 256, blk, 0, stream>>>(ha,     haPh, haPl, 64,  2, AH_DIM);
  pack_b_kernel<<<(16 * 32 * 32) / 256, blk, 0, stream>>>(W1,               W1aPh, W1aPl, 16, 32, Z_DIM, 0);
  pack_b_kernel<<<( 2 * 32 * 32) / 256, blk, 0, stream>>>(W1 + (size_t)Z_DIM * Z_DIM, W1bPh, W1bPl, 2, 32, Z_DIM, 0);
  pack_b_kernel<<<(16 * 32 * 32) / 256, blk, 0, stream>>>(W2,    W2tPh, W2tPl, 16, 32, Z_DIM, 1);
  pack_b_kernel<<<(16 * 64 * 32) / 256, blk, 0, stream>>>(z_hat, zhPh,  zhPl,  16, 64, Z_DIM, 1);

  // U = z @ W1[:512] + b1 ; G = ha @ W1[512:] ; V = zn @ W2^T ; sim = zn @ zhat^T
  gemm_wmma_kernel<<< 64, blk, 0, stream>>>(zPh,  zPl,  W1aPh, W1aPl, U,      b1,      16, 32, Z_DIM);
  gemm_wmma_kernel<<< 64, blk, 0, stream>>>(haPh, haPl, W1bPh, W1bPl, G,      nullptr,  2, 32, Z_DIM);
  gemm_wmma_kernel<<< 64, blk, 0, stream>>>(znPh, znPl, W2tPh, W2tPl, V,      nullptr, 16, 32, Z_DIM);
  gemm_wmma_kernel<<<128, blk, 0, stream>>>(znPh, znPl, zhPh,  zhPl,  simmat, nullptr, 16, 64, NCOL);

  // negative similarities (fused relu-dot over the 99 shifts)
  neg_kernel<<<128, blk, 0, stream>>>(U, G, V, z, z_next, b2, simmat);

  // loss + top-k statistics
  rowstat_kernel<<<128, blk, 0, stream>>>(simmat, row_loss, row_rank);
  reduce_kernel<<<1, blk, 0, stream>>>(row_loss, row_rank, out);
}
